// Attention_76871324664662
// MI455X (gfx1250) — matrix operand order repaired and checked by reading, not compiled, not run on hardware
//
#include <hip/hip_runtime.h>
#include <hip/hip_bf16.h>
#include <math.h>

// ---------------------------------------------------------------------------
// GQA attention layer with RoPE for MI455X (gfx1250, wave32, WMMA).
// All matmuls use v_wmma_f32_16x16x32_bf16 (bf16 in, f32 accumulate).
// GEMM stages B tiles into LDS via async loads (global_load_async_to_lds_b128,
// ASYNCcnt); hot loops are software-pipelined to overlap loads with the
// WMMA pipe. Attention masks only the single tile pair that can touch the
// causal diagonal. All transposes go through LDS tiles so every global
// access is coalesced.
// ---------------------------------------------------------------------------

typedef __attribute__((ext_vector_type(16))) __bf16 v16bf;
typedef __attribute__((ext_vector_type(8)))  __bf16 v8bf;
typedef __attribute__((ext_vector_type(8)))  float  v8f;
typedef __attribute__((ext_vector_type(4)))  float  v4f;

constexpr int BB  = 2;
constexpr int S   = 2048;
constexpr int DM  = 2048;
constexpr int NH  = 32;
constexpr int NKV = 8;
constexpr int DH  = 64;

static __device__ __forceinline__ v16bf cat8(v8bf lo, v8bf hi) {
  return __builtin_shufflevector(lo, hi, 0,1,2,3,4,5,6,7,8,9,10,11,12,13,14,15);
}

static __device__ __forceinline__ v8bf cvt8(const float* __restrict__ p) {
  v4f a = *(const v4f*)p;
  v4f b = *(const v4f*)(p + 4);
  v8bf r;
  r[0] = (__bf16)a[0]; r[1] = (__bf16)a[1]; r[2] = (__bf16)a[2]; r[3] = (__bf16)a[3];
  r[4] = (__bf16)b[0]; r[5] = (__bf16)b[1]; r[6] = (__bf16)b[2]; r[7] = (__bf16)b[3];
  return r;
}

static __device__ __forceinline__ v8f wmma_bf16(v16bf a, v16bf b, v8f c) {
  // (neg_a, A, neg_b, B, c_mod, C, reuse_a, reuse_b)
  return __builtin_amdgcn_wmma_f32_16x16x32_bf16(false, a, false, b, (short)0, c,
                                                 false, false);
}

// Async copy of 32 bytes global -> LDS (two B128 beats; per the ISA the
// instruction offset applies to BOTH the global and LDS addresses).
// Tracked by ASYNCcnt. Low 32 bits of a generic pointer to LDS are the
// wave-relative LDS byte offset (flat-LDS aperture truncation).
static __device__ __forceinline__ void async_cp32(const __bf16* g, __bf16* l) {
  asm volatile("global_load_async_to_lds_b128 %0, %1, off\n\t"
               "global_load_async_to_lds_b128 %0, %1, off offset:16"
               :: "v"((uint32_t)(uintptr_t)l), "v"((uint64_t)(uintptr_t)g)
               : "memory");
}
static __device__ __forceinline__ void wait_async0() {
  asm volatile("s_wait_asynccnt 0x0" ::: "memory");
}

// ---------------------------------------------------------------------------
// Tiled GEMM: C[M,N] = A[M,K] * Bt^T  (Bt stored transposed: [N][K] bf16)
// 128x64 tile per 4-wave workgroup; each wave owns 32 rows x 64 cols.
// The shared 64x32 B tile is staged into LDS once per workgroup with async
// loads (double-buffered; one barrier per k-step). A fragments are loaded
// direct from global (need f32->bf16 convert) and software-pipelined.
// Fragment layouts per CDNA5 ISA:
//   A (16x32 bf16): lane = row, elems = K {klo..klo+7, 16+klo..23+klo}
//   B (32x16 bf16): lane = col, elems = K {klo..klo+7, 16+klo..23+klo}
//   D (16x16 f32) : vgpr = row (+8 upper half), lane = col
// ---------------------------------------------------------------------------
template <bool A_IS_BF16>
__global__ __launch_bounds__(128) void gemm128(const void* __restrict__ Ap,
                                               const __bf16* __restrict__ Bt,
                                               float* __restrict__ C,
                                               int M, int N, int K) {
  __shared__ __align__(32) __bf16 ldsB[2][64 * 32];  // [buf][col*32 + k], 4KB each

  const int lane   = threadIdx.x & 31;
  const int wave   = threadIdx.x >> 5;
  const int l15    = lane & 15;
  const int lanehi = lane >> 4;
  const int klo    = lanehi << 3;   // 0 or 8
  const int khi    = lanehi << 4;   // 0 or 16
  const int m0     = blockIdx.y * 128 + wave * 32;
  const int n0     = blockIdx.x * 64;
  const size_t row0 = (size_t)(m0 + l15);
  const size_t row1 = row0 + 16;

  // Async B staging: thread t covers col = t>>1, 16-element segment t&1.
  const int tcol = threadIdx.x >> 1;
  const int tseg = (threadIdx.x & 1) * 16;
  const __bf16* gB = Bt + (size_t)(n0 + tcol) * K + tseg;
  __bf16* lB = &ldsB[0][tcol * 32 + tseg];

  auto stage = [&](int kk, int buf) { async_cp32(gB + kk, lB + buf * (64 * 32)); };
  auto ldsFragB = [&](int buf, int n) -> v16bf {
    return cat8(*(const v8bf*)&ldsB[buf][(16 * n + l15) * 32 + klo],
                *(const v8bf*)&ldsB[buf][(16 * n + l15) * 32 + 16 + klo]);
  };
  auto loadA = [&](size_t row, int kk) -> v16bf {
    if constexpr (A_IS_BF16) {
      const __bf16* Ar = (const __bf16*)Ap + row * K + kk;
      return cat8(*(const v8bf*)(Ar + klo), *(const v8bf*)(Ar + 16 + klo));
    } else {
      const float* Ar = (const float*)Ap + row * K + kk;
      return cat8(cvt8(Ar + klo), cvt8(Ar + 16 + klo));
    }
  };

  v8f acc[8] = {};

  stage(0, 0);
  v16bf a0 = loadA(row0, 0), a1 = loadA(row1, 0);

  int buf = 0;
  for (int kk = 0; kk < K; kk += 32, buf ^= 1) {
    wait_async0();       // own wave's staged tile is in LDS
    __syncthreads();     // all waves staged; previous buffer's readers done

    if (kk + 32 < K) stage(kk + 32, buf ^ 1);
    const int kn = (kk + 32 < K) ? kk + 32 : kk;  // clamped A prefetch
    v16bf a0n = loadA(row0, kn), a1n = loadA(row1, kn);

    v16bf b0 = ldsFragB(buf, 0), b1 = ldsFragB(buf, 1);
    v16bf b2 = ldsFragB(buf, 2), b3 = ldsFragB(buf, 3);

    acc[0] = wmma_bf16(a0, b0, acc[0]);
    acc[4] = wmma_bf16(a1, b0, acc[4]);
    acc[1] = wmma_bf16(a0, b1, acc[1]);
    acc[5] = wmma_bf16(a1, b1, acc[5]);
    acc[2] = wmma_bf16(a0, b2, acc[2]);
    acc[6] = wmma_bf16(a1, b2, acc[6]);
    acc[3] = wmma_bf16(a0, b3, acc[3]);
    acc[7] = wmma_bf16(a1, b3, acc[7]);

    a0 = a0n; a1 = a1n;
  }

#pragma unroll
  for (int half = 0; half < 2; ++half) {
    const int rbase = m0 + 16 * half + klo;
#pragma unroll
    for (int n = 0; n < 4; ++n) {
      float* Cp = C + (size_t)rbase * N + (n0 + 16 * n + l15);
      const v8f& a = acc[4 * half + n];
#pragma unroll
      for (int r = 0; r < 8; ++r) Cp[(size_t)r * N] = a[r];
    }
  }
}

// ---------------------------------------------------------------------------
// Weight transpose + bf16 convert via LDS tile: Wt[n][k] = bf16(W[k][n]).
// 32x32 tile, 256 threads; reads coalesced along n, writes coalesced along k.
// ---------------------------------------------------------------------------
__global__ __launch_bounds__(256) void w_trans(const float* __restrict__ W,
                                               __bf16* __restrict__ Wt,
                                               int K, int N) {
  __shared__ __bf16 tile[32][33];  // pad to dodge bank conflicts
  const int k0 = blockIdx.x * 32;
  const int n0 = blockIdx.y * 32;
  const int tx = threadIdx.x & 31;
  const int ty = threadIdx.x >> 5;  // 0..7
#pragma unroll
  for (int i = 0; i < 32; i += 8)
    tile[ty + i][tx] = (__bf16)W[(size_t)(k0 + ty + i) * N + (n0 + tx)];
  __syncthreads();
#pragma unroll
  for (int i = 0; i < 32; i += 8)
    Wt[(size_t)(n0 + ty + i) * K + (k0 + tx)] = tile[tx][ty + i];
}

// ---------------------------------------------------------------------------
// RoPE for Q: qf [B,S,NH*64] (f32) -> qb [B,NH,S,64] (bf16, roped)
// ---------------------------------------------------------------------------
__global__ void rope_q(const float* __restrict__ qf,
                       const float* __restrict__ cosT,
                       const float* __restrict__ sinT,
                       __bf16* __restrict__ qb) {
  int idx = blockIdx.x * 256 + threadIdx.x;  // over S*32
  int s = idx >> 5, d2 = idx & 31;
  int h = blockIdx.y, b = blockIdx.z;
  const float* row = qf + ((size_t)(b * S + s)) * (NH * DH) + h * DH;
  float x1 = row[d2], x2 = row[d2 + 32];
  float c1 = cosT[s * DH + d2],      s1 = sinT[s * DH + d2];
  float c2 = cosT[s * DH + d2 + 32], s2 = sinT[s * DH + d2 + 32];
  __bf16* orow = qb + (((size_t)(b * NH + h)) * S + s) * DH;
  orow[d2]      = (__bf16)(x1 * c1 - x2 * s1);
  orow[d2 + 32] = (__bf16)(x2 * c2 + x1 * s2);
}

// ---------------------------------------------------------------------------
// RoPE for K: kf [B,S,NKV*64] (f32) -> new_k [B,NKV,S,64] (f32, to d_out)
//                                       kb   [B,NKV,S,64] (bf16)
// ---------------------------------------------------------------------------
__global__ void rope_k(const float* __restrict__ kf,
                       const float* __restrict__ cosT,
                       const float* __restrict__ sinT,
                       float* __restrict__ newk, __bf16* __restrict__ kb) {
  int idx = blockIdx.x * 256 + threadIdx.x;  // over S*32
  int s = idx >> 5, d2 = idx & 31;
  int h = blockIdx.y, b = blockIdx.z;
  const float* row = kf + ((size_t)(b * S + s)) * (NKV * DH) + h * DH;
  float x1 = row[d2], x2 = row[d2 + 32];
  float c1 = cosT[s * DH + d2],      s1 = sinT[s * DH + d2];
  float c2 = cosT[s * DH + d2 + 32], s2 = sinT[s * DH + d2 + 32];
  float o1 = x1 * c1 - x2 * s1;
  float o2 = x2 * c2 + x1 * s2;
  size_t ob = (((size_t)(b * NKV + h)) * S + s) * DH;
  newk[ob + d2]      = o1;
  newk[ob + d2 + 32] = o2;
  kb[ob + d2]        = (__bf16)o1;
  kb[ob + d2 + 32]   = (__bf16)o2;
}

// ---------------------------------------------------------------------------
// V layout via LDS tile: vf [B,S,NKV*64] (f32) ->
//   new_v [B,NKV,S,64] (f32, to d_out, coalesced along d)
//   vt    [B,NKV,64,S] (bf16, transposed; written coalesced along s)
// 64x64 (s x d) tile per 256-thread block.
// ---------------------------------------------------------------------------
__global__ __launch_bounds__(256) void v_layout(const float* __restrict__ vf,
                                                float* __restrict__ newv,
                                                __bf16* __restrict__ vt) {
  __shared__ __bf16 tile[64][65];  // [s_local][d], padded
  const int s0 = blockIdx.x * 64;
  const int h = blockIdx.y, b = blockIdx.z;
  const int tx = threadIdx.x & 63;
  const int ty = threadIdx.x >> 6;  // 0..3
#pragma unroll
  for (int i = 0; i < 64; i += 4) {
    const int sl = ty + i;
    float val = vf[((size_t)(b * S + s0 + sl)) * (NKV * DH) + h * DH + tx];
    newv[(((size_t)(b * NKV + h)) * S + (s0 + sl)) * DH + tx] = val;
    tile[sl][tx] = (__bf16)val;
  }
  __syncthreads();
#pragma unroll
  for (int i = 0; i < 64; i += 4) {
    const int dl = ty + i;
    vt[(((size_t)(b * NKV + h)) * DH + dl) * S + (s0 + tx)] = tile[tx][dl];
  }
}

// ---------------------------------------------------------------------------
// Flash-style causal GQA attention. 4 waves / block, one 16-row q-tile per
// wave. Per 32-key tile:
//   S^T(t,q) = K_tile x Q^T   (softmax stats mostly in-lane over acc VGPRs)
//   O^T(d,q) = V^T x P^T      (P^T repacked from S^T D-frag with xor-16 shfls)
// Only the final tile pair can intersect the causal diagonal, so the main
// loop carries no mask compares at all; one masked epilogue iteration
// consumes the K fragments left in the software pipeline.
// Softmax runs in the log2 domain (raw v_exp_f32, scale folded in).
// ---------------------------------------------------------------------------
__global__ __launch_bounds__(128) void attn_kernel(const __bf16* __restrict__ Q,
                                                   const __bf16* __restrict__ Kb,
                                                   const __bf16* __restrict__ Vt,
                                                   __bf16* __restrict__ att) {
  const int wave = threadIdx.x >> 5;
  const int qt = blockIdx.x * 4 + wave;  // q tile (16 rows)
  const int h  = blockIdx.y;             // query head
  const int b  = blockIdx.z;
  const int g  = h >> 2;                 // kv head (N_REP = 4)
  const int lane   = threadIdx.x & 31;
  const int l15    = lane & 15;
  const int lanehi = lane >> 4;
  const int klo    = lanehi << 3;
  const int khi    = lanehi << 4;
  const int qrow   = qt * 16 + l15;

  const __bf16* Qh = Q  + ((size_t)(b * NH  + h)) * S  * DH;
  const __bf16* Kh = Kb + ((size_t)(b * NKV + g)) * S  * DH;
  const __bf16* Vh = Vt + ((size_t)(b * NKV + g)) * DH * S;

  // Q^T B-fragments: lane = q column, elems = d {klo..klo+7, 16+klo..23+klo} per 32-d half
  const __bf16* qp = Qh + (size_t)qrow * DH;
  v16bf bq0 = cat8(*(const v8bf*)(qp + klo), *(const v8bf*)(qp + 16 + klo));        // d =  0..31
  v16bf bq1 = cat8(*(const v8bf*)(qp + 32 + klo), *(const v8bf*)(qp + 48 + klo));   // d = 32..63

  auto loadK = [&](int tbase, v16bf k[4]) {
    const __bf16* kp0 = Kh + (size_t)(tbase + l15) * DH;
    const __bf16* kp1 = kp0 + 16 * DH;
    k[0] = cat8(*(const v8bf*)(kp0 + klo),      *(const v8bf*)(kp0 + 16 + klo));
    k[1] = cat8(*(const v8bf*)(kp0 + 32 + klo), *(const v8bf*)(kp0 + 48 + klo));
    k[2] = cat8(*(const v8bf*)(kp1 + klo),      *(const v8bf*)(kp1 + 16 + klo));
    k[3] = cat8(*(const v8bf*)(kp1 + 32 + klo), *(const v8bf*)(kp1 + 48 + klo));
  };
  auto loadV = [&](int tbase, v16bf v[4]) {
#pragma unroll
    for (int j = 0; j < 4; ++j) {
      const __bf16* vp = Vh + (size_t)(16 * j + l15) * S + tbase;
      v[j] = cat8(*(const v8bf*)(vp + khi), *(const v8bf*)(vp + khi + 8));
    }
  };

  v8f acc[4] = {};                 // O^T accumulators, d-tiles of 16
  float mrun = -INFINITY, lrun = 0.0f;
  const float sc2 = 0.125f * 1.44269504088896340736f;  // (1/sqrt(64)) * log2(e)

  // Tiles with tbase+31 <= qbase are fully unmasked; exactly one masked
  // tile pair remains (the last one).
  const int qb16   = qt * 16;
  const int nfull  = (qt >= 2) ? ((qb16 - 31) >> 5) + 1 : 0;

  v16bf kc[4], kn[4], vc[4];
  loadK(0, kc);

  // ---- main loop: no causal compares, prefetch always in-bounds ----
  for (int tt = 0; tt < nfull; ++tt) {
    const int tbase = tt * 32;
    loadV(tbase, vc);
    loadK(tbase + 32, kn);

    v8f st0 = {}, st1 = {};
    st0 = wmma_bf16(kc[0], bq0, st0);
    st1 = wmma_bf16(kc[2], bq0, st1);
    st0 = wmma_bf16(kc[1], bq1, st0);
    st1 = wmma_bf16(kc[3], bq1, st1);

    float mt = st0[0] * sc2;
#pragma unroll
    for (int r = 0; r < 8; ++r) {
      st0[r] *= sc2;
      st1[r] *= sc2;
      mt = fmaxf(mt, fmaxf(st0[r], st1[r]));
    }
    mt = fmaxf(mt, __shfl_xor(mt, 16, 32));
    const float mnew = fmaxf(mrun, mt);
    const float corr = exp2f(mrun - mnew);
    lrun *= corr;
#pragma unroll
    for (int j = 0; j < 4; ++j)
#pragma unroll
      for (int r = 0; r < 8; ++r) acc[j][r] *= corr;

    v16bf bp;
    float lsum = 0.0f;
#pragma unroll
    for (int r = 0; r < 8; ++r) {
      float p0 = exp2f(st0[r] - mnew);
      float p1 = exp2f(st1[r] - mnew);
      lsum += p0 + p1;
      float x0 = __shfl_xor(p0, 16, 32);
      float x1 = __shfl_xor(p1, 16, 32);
      bp[r]     = (__bf16)(lanehi ? x1 : p0);
      bp[8 + r] = (__bf16)(lanehi ? p1 : x0);
    }
    lsum += __shfl_xor(lsum, 16, 32);
    lrun += lsum;
    mrun = mnew;

#pragma unroll
    for (int j = 0; j < 4; ++j) acc[j] = wmma_bf16(vc[j], bp, acc[j]);
#pragma unroll
    for (int j = 0; j < 4; ++j) kc[j] = kn[j];
  }

  // ---- masked epilogue: the single diagonal tile pair (kc already loaded) ----
  {
    const int tbase = nfull * 32;
    loadV(tbase, vc);

    v8f st0 = {}, st1 = {};
    st0 = wmma_bf16(kc[0], bq0, st0);
    st1 = wmma_bf16(kc[2], bq0, st1);
    st0 = wmma_bf16(kc[1], bq1, st0);
    st1 = wmma_bf16(kc[3], bq1, st1);

    float mt = -INFINITY;
#pragma unroll
    for (int r = 0; r < 8; ++r) {
      int t0 = tbase + klo + r;
      st0[r] = (t0 <= qrow) ? st0[r] * sc2 : -INFINITY;
      int t1 = t0 + 16;
      st1[r] = (t1 <= qrow) ? st1[r] * sc2 : -INFINITY;
      mt = fmaxf(mt, fmaxf(st0[r], st1[r]));
    }
    mt = fmaxf(mt, __shfl_xor(mt, 16, 32));
    const float mnew = fmaxf(mrun, mt);
    const float corr = exp2f(mrun - mnew);
    lrun *= corr;
#pragma unroll
    for (int j = 0; j < 4; ++j)
#pragma unroll
      for (int r = 0; r < 8; ++r) acc[j][r] *= corr;

    v16bf bp;
    float lsum = 0.0f;
#pragma unroll
    for (int r = 0; r < 8; ++r) {
      float p0 = exp2f(st0[r] - mnew);
      float p1 = exp2f(st1[r] - mnew);
      lsum += p0 + p1;
      float x0 = __shfl_xor(p0, 16, 32);
      float x1 = __shfl_xor(p1, 16, 32);
      bp[r]     = (__bf16)(lanehi ? x1 : p0);
      bp[8 + r] = (__bf16)(lanehi ? p1 : x0);
    }
    lsum += __shfl_xor(lsum, 16, 32);
    lrun += lsum;

#pragma unroll
    for (int j = 0; j < 4; ++j) acc[j] = wmma_bf16(vc[j], bp, acc[j]);
  }

  // Normalize and store O^T packed: vgpr r -> d = 16j + klo + r (contiguous
  // per lane), lane -> q = qrow. One 16B store per d-tile instead of 8x2B.
  const float inv = 1.0f / lrun;
  __bf16* ap = att + ((size_t)(b * S) + qrow) * (NH * DH) + h * DH;
#pragma unroll
  for (int j = 0; j < 4; ++j) {
    v8bf o;
#pragma unroll
    for (int r = 0; r < 8; ++r) o[r] = (__bf16)(acc[j][r] * inv);
    *(v8bf*)(ap + 16 * j + klo) = o;
  }
}

// ---------------------------------------------------------------------------
// Host-side orchestration (graph-capture safe: only kernel launches on stream)
// ---------------------------------------------------------------------------
extern "C" void kernel_launch(void* const* d_in, const int* in_sizes, int n_in,
                              void* d_out, int out_size, void* d_ws, size_t ws_size,
                              hipStream_t stream) {
  (void)in_sizes; (void)n_in; (void)out_size; (void)ws_size;

  const float* x    = (const float*)d_in[0];
  const float* cosT = (const float*)d_in[1];
  const float* sinT = (const float*)d_in[2];
  const float* Wq   = (const float*)d_in[3];
  const float* Wk   = (const float*)d_in[4];
  const float* Wv   = (const float*)d_in[5];
  const float* Wo   = (const float*)d_in[6];

  float* out  = (float*)d_out;                         // [B,S,D]
  float* newk = out + (size_t)BB * S * DM;             // [B,NKV,S,64]
  float* newv = newk + (size_t)BB * NKV * S * DH;      // [B,NKV,S,64]

  // Workspace carve (256B aligned), with aliasing of dead regions:
  //   qf region reused for att; kf region reused for Wo^T.
  char* w = (char*)d_ws;
  auto take = [&](size_t bytes) -> char* {
    char* p = w;
    w += (bytes + 255) & ~(size_t)255;
    return p;
  };
  float*  qf  = (float*)take((size_t)BB * S * NH  * DH * 4);  // 33.5 MB
  float*  kf  = (float*)take((size_t)BB * S * NKV * DH * 4);  //  8.4 MB
  float*  vf  = (float*)take((size_t)BB * S * NKV * DH * 4);  //  8.4 MB
  __bf16* wtq = (__bf16*)take((size_t)(NH * DH) * DM * 2);    //  8.4 MB
  __bf16* wtk = (__bf16*)take((size_t)(NKV * DH) * DM * 2);   //  2.1 MB
  __bf16* wtv = (__bf16*)take((size_t)(NKV * DH) * DM * 2);   //  2.1 MB
  __bf16* qb  = (__bf16*)take((size_t)BB * NH  * S * DH * 2); // 16.8 MB
  __bf16* kb  = (__bf16*)take((size_t)BB * NKV * S * DH * 2); //  4.2 MB
  __bf16* vt  = (__bf16*)take((size_t)BB * NKV * S * DH * 2); //  4.2 MB
  __bf16* att = (__bf16*)qf;   // alias: qf dead after rope_q
  __bf16* wto = (__bf16*)kf;   // alias: kf dead after rope_k

  const int M = BB * S;  // 4096 token rows

  // 1) Weight transpose+convert for QKV (LDS-tiled, fully coalesced)
  w_trans<<<dim3(DM / 32, (NH * DH) / 32),  256, 0, stream>>>(Wq, wtq, DM, NH * DH);
  w_trans<<<dim3(DM / 32, (NKV * DH) / 32), 256, 0, stream>>>(Wk, wtk, DM, NKV * DH);
  w_trans<<<dim3(DM / 32, (NKV * DH) / 32), 256, 0, stream>>>(Wv, wtv, DM, NKV * DH);

  // 2) QKV projections (f32 activations -> bf16 frags -> f32 results)
  gemm128<false><<<dim3((NH * DH) / 64,  M / 128), 128, 0, stream>>>(x, wtq, qf, M, NH * DH,  DM);
  gemm128<false><<<dim3((NKV * DH) / 64, M / 128), 128, 0, stream>>>(x, wtk, kf, M, NKV * DH, DM);
  gemm128<false><<<dim3((NKV * DH) / 64, M / 128), 128, 0, stream>>>(x, wtv, vf, M, NKV * DH, DM);

  // 3) RoPE + layout transforms (also emit new_k / new_v f32 to d_out)
  rope_q  <<<dim3((S * 32) / 256, NH,  BB), 256, 0, stream>>>(qf, cosT, sinT, qb);
  rope_k  <<<dim3((S * 32) / 256, NKV, BB), 256, 0, stream>>>(kf, cosT, sinT, newk, kb);
  v_layout<<<dim3(S / 64, NKV, BB), 256, 0, stream>>>(vf, newv, vt);

  // 4) Wo transpose (into region freed by kf)
  w_trans<<<dim3((NH * DH) / 32, DM / 32), 256, 0, stream>>>(Wo, wto, NH * DH, DM);

  // 5) Flash attention (att written into region freed by qf)
  attn_kernel<<<dim3(S / 64, NH, BB), 128, 0, stream>>>(qb, kb, vt, att);

  // 6) Output projection
  gemm128<true><<<dim3(DM / 64, M / 128), 128, 0, stream>>>(att, wto, out, M, DM, NH * DH);
}
